// Bottleneck_52201032516128
// MI455X (gfx1250) — matrix operand order repaired and checked by reading, not compiled, not run on hardware
//
#include <hip/hip_runtime.h>
#include <hip/hip_bf16.h>
#include <cstdint>

// ---------------------------------------------------------------------------
// MI455X (gfx1250) implementation.
// bf16 WMMA (v_wmma_f32_16x16x32_bf16) for the three convs, f32 accumulate,
// masks/bias/ReLU fused in epilogues. A-tiles staged into LDS by the Tensor
// Data Mover (tensor_load_to_lds, TENSORcnt) when the builtin is available;
// register staging fallback otherwise. Working set (~130MB) is L2-resident
// (192MB), so HBM traffic ~= read x + write out ~= 206MB @ 23.3TB/s.
// ---------------------------------------------------------------------------

typedef __attribute__((ext_vector_type(16))) __bf16 v16bf;
typedef __attribute__((ext_vector_type(8)))  __bf16 v8bf;
typedef __attribute__((ext_vector_type(8)))  float  v8f;
typedef __attribute__((ext_vector_type(4)))  unsigned int u32x4;
typedef __attribute__((ext_vector_type(8)))  int i32x8;
typedef __attribute__((ext_vector_type(4)))  int i32x4;

#define B_   16
#define C_   512
#define WID_ 128
#define HW_  3136      // 56*56
#define M_   (B_*HW_)  // 50176

#if __has_builtin(__builtin_amdgcn_tensor_load_to_lds) && \
    __has_builtin(__builtin_amdgcn_s_wait_tensorcnt)
#define HAVE_TDM 1
#else
#define HAVE_TDM 0
#endif

__device__ __forceinline__ v8f wmma_bf16(v16bf a, v16bf b, v8f c) {
  return __builtin_amdgcn_wmma_f32_16x16x32_bf16(false, a, false, b, (short)0, c,
                                                 false, false);
}

// A-fragment in contiguous K order (matches B): lane-half lh holds
// K = {lh*16..lh*16+15}. Gather from an unswizzled row.
__device__ __forceinline__ v16bf load_a_frag(const __bf16* rowp, int lh) {
  union { v16bf v; v8bf h[2]; } u;
  u.h[0] = *(const v8bf*)(rowp + lh * 16);
  u.h[1] = *(const v8bf*)(rowp + lh * 16 + 8);
  return u.v;
}

#if HAVE_TDM
// TDM 2D tile load: 128 rows x 32 bf16, row stride = krow elements, into LDS.
// D# layout per CDNA5 ISA.
__device__ __forceinline__ void tdm_load_a_tile(const __bf16* gsrc,
                                                unsigned int lds_byte, int krow) {
  unsigned long long ga = (unsigned long long)(size_t)gsrc;
  u32x4 g0;
  g0[0] = 1u;                                             // count=1, user mode
  g0[1] = lds_byte;                                       // lds_addr
  g0[2] = (unsigned int)ga;                               // global_addr[31:0]
  g0[3] = (unsigned int)((ga >> 32) & 0x01FFFFFFull) | (2u << 30);  // type=2
  i32x8 g1;
  g1[0] = 1 << 16;                                        // data_size=1 -> 2B
  g1[1] = (int)((unsigned int)krow << 16);                // tensor_dim0 lo16
  g1[2] = (int)((((unsigned int)krow >> 16) & 0xFFFFu) | (128u << 16)); // td0 hi | td1 lo
  g1[3] = (int)(32u << 16);                               // td1 hi=0 | tile_dim0=32
  g1[4] = 128;                                            // tile_dim1=128, tile_dim2=0
  g1[5] = krow;                                           // tensor_dim0_stride lo32
  g1[6] = 0;                                              // stride hi | td1_stride lo
  g1[7] = 0;
  i32x4 gz = {0, 0, 0, 0};                                // 2D: groups 2/3 unused
#if __clang_major__ >= 23
  i32x8 gz8 = {0, 0, 0, 0, 0, 0, 0, 0};
  __builtin_amdgcn_tensor_load_to_lds(g0, g1, gz, gz, gz8, 0);
#else
  __builtin_amdgcn_tensor_load_to_lds(g0, g1, gz, gz, 0);
#endif
}
#endif

// ---------------- prep kernels ----------------

__global__ void cvt_bf16_k(const float* __restrict__ s, __bf16* __restrict__ d, int n) {
  int i = blockIdx.x * 256 + threadIdx.x;
  if (i < n) d[i] = (__bf16)s[i];
}

// w2 (OIHW 128,128,3,3) -> [tap][n][k] bf16
__global__ void perm_w2_k(const float* __restrict__ w2, __bf16* __restrict__ d) {
  int i = blockIdx.x * 256 + threadIdx.x;          // < 9*128*128
  int t = i >> 14, r = i & 16383;
  int n = r >> 7, k = r & 127;
  int ky = t / 3, kx = t % 3;
  d[i] = (__bf16)w2[(((size_t)n * 128 + k) * 3 + ky) * 3 + kx];
}

// NCHW f32 -> NHWC bf16 via 32x32 LDS tiles
__global__ __launch_bounds__(256) void transpose_cvt_k(const float* __restrict__ X,
                                                       __bf16* __restrict__ XT) {
  __shared__ float tile[32][33];
  int b = blockIdx.z;
  int p0 = blockIdx.x * 32, c0 = blockIdx.y * 32;
  int tc = threadIdx.x & 31, tr = threadIdx.x >> 5;
  const float* xb = X + (size_t)b * C_ * HW_;
  for (int i = 0; i < 4; ++i)
    tile[tr + i * 8][tc] = xb[(size_t)(c0 + tr + i * 8) * HW_ + p0 + tc];
  __syncthreads();
  __bf16* xt = XT + (size_t)b * HW_ * C_;
  for (int i = 0; i < 4; ++i)
    xt[(size_t)(p0 + tr + i * 8) * C_ + c0 + tc] = (__bf16)tile[tc][tr + i * 8];
}

__global__ void zero_k(unsigned int* __restrict__ p, int nwords) {
  int i = blockIdx.x * 256 + threadIdx.x;
  if (i < nwords) p[i] = 0u;
}

// s_mask: per-pixel 512-dot with sm_w (from bf16 NHWC copy)
__global__ void smask_k(const __bf16* __restrict__ XT, const float* __restrict__ smw,
                        const float* __restrict__ smb, float* __restrict__ SM) {
  int gp = blockIdx.x * 256 + threadIdx.x;  // < M_
  const __bf16* row = XT + (size_t)gp * C_;
  float s = 0.f;
  for (int k = 0; k < C_; ++k) s += (float)row[k] * smw[k];
  SM[gp] = (s + smb[0]) > 0.f ? 1.f : 0.f;
}

// 3x3 max dilation
__global__ void dilate_k(const float* __restrict__ SM, float* __restrict__ SMD) {
  int gp = blockIdx.x * 256 + threadIdx.x;
  int b = gp / HW_, p = gp % HW_, y = p / 56, x = p % 56;
  float m = 0.f;
  for (int dy = -1; dy <= 1; ++dy)
    for (int dx = -1; dx <= 1; ++dx) {
      int yy = y + dy, xx = x + dx;
      if (yy >= 0 && yy < 56 && xx >= 0 && xx < 56)
        m = fmaxf(m, SM[b * HW_ + yy * 56 + xx]);
    }
  SMD[gp] = m;
}

// per-(b,c) spatial mean of x (NCHW, contiguous rows) — one wave per row
__global__ __launch_bounds__(256) void pool_x_k(const float* __restrict__ X,
                                                float* __restrict__ PX) {
  int r = blockIdx.x * 8 + (threadIdx.x >> 5);  // (b,c) row < 8192
  int lane = threadIdx.x & 31;
  const float* row = X + (size_t)r * HW_;
  float s = 0.f;
  for (int i = lane; i < HW_; i += 32) s += row[i];
  for (int o = 16; o > 0; o >>= 1) s += __shfl_xor(s, o, 32);
  if (lane == 0) PX[r] = s * (1.f / HW_);
}

// channel mask: mask[b,n] = (pooled[b,:] . W[n,:] + bias[n]) > 0
__global__ void cmask_k(const float* __restrict__ P, const float* __restrict__ W,
                        const float* __restrict__ bias, float* __restrict__ Mk,
                        int N, int K) {
  int i = blockIdx.x * 256 + threadIdx.x;
  int b = i / N, n = i % N;
  const float* p = P + (size_t)b * K;
  const float* w = W + (size_t)n * K;
  float s = bias[n];
  for (int k = 0; k < K; ++k) s += p[k] * w[k];
  Mk[i] = s > 0.f ? 1.f : 0.f;
}

// ---------------- GEMM 1: h1 = relu(xT @ w1^T + b1) * cm1 * smd ----------------
// M=50176 K=512 N=128. Block tile 128x64, 8 waves (4Mx2N), wave 32x32.
// A-tile (128x32 bf16) staged by TDM; B-tile by register staging.
__global__ __launch_bounds__(256) void gemm1_k(const __bf16* __restrict__ A,
                                               const __bf16* __restrict__ Bw,
                                               const float* __restrict__ bias,
                                               const float* __restrict__ cm,
                                               const float* __restrict__ smd,
                                               __bf16* __restrict__ H1P) {
  __shared__ __bf16 As[2][128][32];
  __shared__ __bf16 Bs[2][64][32];
  const int t = threadIdx.x, lane = t & 31, wid = t >> 5;
  const int mw = wid >> 1, nw = wid & 1;
  const int m0 = blockIdx.y * 128, n0 = blockIdx.x * 64;
  const int lm = lane & 15, lh = lane >> 4;
  const int brow = t >> 2, bkq = t & 3;
#if HAVE_TDM
  const unsigned int ldsA[2] = {(unsigned int)(size_t)&As[0][0][0],
                                (unsigned int)(size_t)&As[1][0][0]};
#else
  const int arow = t >> 1, akh = t & 1;
#endif

  v8f acc[2][2] = {};

  // stage 0 (A unswizzled; fragment gather reads contiguous K)
#if HAVE_TDM
  if (wid == 0) tdm_load_a_tile(A + (size_t)m0 * 512, ldsA[0], 512);
#else
  *(v16bf*)&As[0][arow][akh * 16] =
      *(const v16bf*)&A[(size_t)(m0 + arow) * 512 + akh * 16];
#endif
  *(v8bf*)&Bs[0][brow][bkq * 8] = *(const v8bf*)&Bw[(size_t)(n0 + brow) * 512 + bkq * 8];
#if HAVE_TDM
  if (wid == 0) __builtin_amdgcn_s_wait_tensorcnt(0);
#endif
  __syncthreads();

  int buf = 0;
  for (int kt = 0; kt < 16; ++kt) {
    if (kt + 1 < 16) {
      int k0 = (kt + 1) * 32;
#if HAVE_TDM
      if (wid == 0) tdm_load_a_tile(A + (size_t)m0 * 512 + k0, ldsA[buf ^ 1], 512);
#else
      __builtin_prefetch(&A[(size_t)(m0 + arow) * 512 + k0 + 32], 0, 1);
      *(v16bf*)&As[buf ^ 1][arow][akh * 16] =
          *(const v16bf*)&A[(size_t)(m0 + arow) * 512 + k0 + akh * 16];
#endif
      *(v8bf*)&Bs[buf ^ 1][brow][bkq * 8] =
          *(const v8bf*)&Bw[(size_t)(n0 + brow) * 512 + k0 + bkq * 8];
    }
    v16bf a0 = load_a_frag(&As[buf][mw * 32 + lm][0], lh);
    v16bf a1 = load_a_frag(&As[buf][mw * 32 + 16 + lm][0], lh);
    v16bf b0 = *(const v16bf*)&Bs[buf][nw * 32 + lm][lh * 16];
    v16bf b1 = *(const v16bf*)&Bs[buf][nw * 32 + 16 + lm][lh * 16];
    acc[0][0] = wmma_bf16(a0, b0, acc[0][0]);
    acc[0][1] = wmma_bf16(a0, b1, acc[0][1]);
    acc[1][0] = wmma_bf16(a1, b0, acc[1][0]);
    acc[1][1] = wmma_bf16(a1, b1, acc[1][1]);
#if HAVE_TDM
    if (wid == 0) __builtin_amdgcn_s_wait_tensorcnt(0);
#endif
    __syncthreads();
    buf ^= 1;
  }

  for (int mt = 0; mt < 2; ++mt)
    for (int nt = 0; nt < 2; ++nt) {
      int gn = n0 + nw * 32 + nt * 16 + lm;
      float cb = bias[gn];
      for (int r = 0; r < 8; ++r) {
        int gm = m0 + mw * 32 + mt * 16 + lh * 8 + r;
        int b = gm / HW_, p = gm - b * HW_;
        int y = p / 56, x = p - y * 56;
        float v = acc[mt][nt][r] + cb;
        v = fmaxf(v, 0.f) * cm[b * WID_ + gn] * smd[gm];
        H1P[(((size_t)(b * 58 + y + 1)) * 58 + (x + 1)) * WID_ + gn] = (__bf16)v;
      }
    }
}

// mean of masked h1 (padded interior) per (b,c)
__global__ __launch_bounds__(128) void pool_h1_k(const __bf16* __restrict__ H1P,
                                                 float* __restrict__ PH) {
  int b = blockIdx.x, c = threadIdx.x;
  const __bf16* hb = H1P + (size_t)b * 58 * 58 * WID_;
  float s = 0.f;
  for (int y = 0; y < 56; ++y) {
    const __bf16* rowp = hb + ((size_t)(y + 1) * 58 + 1) * WID_;
    for (int x = 0; x < 56; ++x) s += (float)rowp[x * WID_ + c];
  }
  PH[b * WID_ + c] = s * (1.f / HW_);
}

// ---------------- GEMM 2: implicit-GEMM 3x3, h2 = relu(conv3(h1)+b2)*cm2*sm ---
// per-batch: M=3136 K=9*128 N=128; block tile 64x128, 8 waves (2Mx4N), wave 32x32
__global__ __launch_bounds__(256) void conv3x3_k(const __bf16* __restrict__ H1P,
                                                 const __bf16* __restrict__ W2B,
                                                 const float* __restrict__ bias,
                                                 const float* __restrict__ cm,
                                                 const float* __restrict__ sm,
                                                 __bf16* __restrict__ H2) {
  const int t = threadIdx.x, lane = t & 31, wid = t >> 5;
  const int mw = wid >> 2, nw = wid & 3;
  const int b = blockIdx.y, m0 = blockIdx.x * 64;
  const int lm = lane & 15, lh = lane >> 4;

  int py[2], px[2];
  for (int mt = 0; mt < 2; ++mt) {
    int p = m0 + mw * 32 + mt * 16 + lm;
    py[mt] = p / 56;
    px[mt] = p - py[mt] * 56;
  }
  const __bf16* hb = H1P + (size_t)b * 58 * 58 * WID_;
  v8f acc[2][2] = {};

  for (int tap = 0; tap < 9; ++tap) {
    int dy = tap / 3 - 1, dx = tap % 3 - 1;
    const __bf16* wt = W2B + (size_t)tap * WID_ * WID_;
    __builtin_prefetch(wt + WID_ * WID_, 0, 2);  // next tap's weights
    const __bf16* ab0 = hb + ((size_t)(py[0] + 1 + dy) * 58 + (px[0] + 1 + dx)) * WID_;
    const __bf16* ab1 = hb + ((size_t)(py[1] + 1 + dy) * 58 + (px[1] + 1 + dx)) * WID_;
    for (int kq = 0; kq < 4; ++kq) {
      int k0 = kq * 32;
      v16bf a0 = load_a_frag(ab0 + k0, lh);
      v16bf a1 = load_a_frag(ab1 + k0, lh);
      for (int nt = 0; nt < 2; ++nt) {
        int n = nw * 32 + nt * 16 + lm;
        v16bf bf = *(const v16bf*)&wt[(size_t)n * WID_ + k0 + lh * 16];
        acc[0][nt] = wmma_bf16(a0, bf, acc[0][nt]);
        acc[1][nt] = wmma_bf16(a1, bf, acc[1][nt]);
      }
    }
  }

  for (int mt = 0; mt < 2; ++mt)
    for (int nt = 0; nt < 2; ++nt) {
      int n = nw * 32 + nt * 16 + lm;
      float cb = bias[n];
      for (int r = 0; r < 8; ++r) {
        int p = m0 + mw * 32 + mt * 16 + lh * 8 + r;
        float v = acc[mt][nt][r] + cb;
        v = fmaxf(v, 0.f) * cm[b * WID_ + n] * sm[b * HW_ + p];
        H2[((size_t)b * HW_ + p) * WID_ + n] = (__bf16)v;
      }
    }
}

__global__ __launch_bounds__(128) void pool_h2_k(const __bf16* __restrict__ H2,
                                                 float* __restrict__ PH) {
  int b = blockIdx.x, c = threadIdx.x;
  const __bf16* hb = H2 + (size_t)b * HW_ * WID_;
  float s = 0.f;
  for (int p = 0; p < HW_; ++p) s += (float)hb[(size_t)p * WID_ + c];
  PH[b * WID_ + c] = s * (1.f / HW_);
}

// ---------------- GEMM 3: h3 = (h2 @ w3^T + b3) * cm3 * sm  (no relu) --------
// M=50176 K=128 N=512. Same scheme as GEMM1, KT=4.
__global__ __launch_bounds__(256) void gemm3_k(const __bf16* __restrict__ A,
                                               const __bf16* __restrict__ Bw,
                                               const float* __restrict__ bias,
                                               const float* __restrict__ cm,
                                               const float* __restrict__ sm,
                                               __bf16* __restrict__ H3) {
  __shared__ __bf16 As[2][128][32];
  __shared__ __bf16 Bs[2][64][32];
  const int t = threadIdx.x, lane = t & 31, wid = t >> 5;
  const int mw = wid >> 1, nw = wid & 1;
  const int m0 = blockIdx.y * 128, n0 = blockIdx.x * 64;
  const int lm = lane & 15, lh = lane >> 4;
  const int brow = t >> 2, bkq = t & 3;
#if HAVE_TDM
  const unsigned int ldsA[2] = {(unsigned int)(size_t)&As[0][0][0],
                                (unsigned int)(size_t)&As[1][0][0]};
#else
  const int arow = t >> 1, akh = t & 1;
#endif

  v8f acc[2][2] = {};
#if HAVE_TDM
  if (wid == 0) tdm_load_a_tile(A + (size_t)m0 * 128, ldsA[0], 128);
#else
  *(v16bf*)&As[0][arow][akh * 16] =
      *(const v16bf*)&A[(size_t)(m0 + arow) * 128 + akh * 16];
#endif
  *(v8bf*)&Bs[0][brow][bkq * 8] = *(const v8bf*)&Bw[(size_t)(n0 + brow) * 128 + bkq * 8];
#if HAVE_TDM
  if (wid == 0) __builtin_amdgcn_s_wait_tensorcnt(0);
#endif
  __syncthreads();

  int buf = 0;
  for (int kt = 0; kt < 4; ++kt) {
    if (kt + 1 < 4) {
      int k0 = (kt + 1) * 32;
#if HAVE_TDM
      if (wid == 0) tdm_load_a_tile(A + (size_t)m0 * 128 + k0, ldsA[buf ^ 1], 128);
#else
      *(v16bf*)&As[buf ^ 1][arow][akh * 16] =
          *(const v16bf*)&A[(size_t)(m0 + arow) * 128 + k0 + akh * 16];
#endif
      *(v8bf*)&Bs[buf ^ 1][brow][bkq * 8] =
          *(const v8bf*)&Bw[(size_t)(n0 + brow) * 128 + k0 + bkq * 8];
    }
    v16bf a0 = load_a_frag(&As[buf][mw * 32 + lm][0], lh);
    v16bf a1 = load_a_frag(&As[buf][mw * 32 + 16 + lm][0], lh);
    v16bf b0 = *(const v16bf*)&Bs[buf][nw * 32 + lm][lh * 16];
    v16bf b1 = *(const v16bf*)&Bs[buf][nw * 32 + 16 + lm][lh * 16];
    acc[0][0] = wmma_bf16(a0, b0, acc[0][0]);
    acc[0][1] = wmma_bf16(a0, b1, acc[0][1]);
    acc[1][0] = wmma_bf16(a1, b0, acc[1][0]);
    acc[1][1] = wmma_bf16(a1, b1, acc[1][1]);
#if HAVE_TDM
    if (wid == 0) __builtin_amdgcn_s_wait_tensorcnt(0);
#endif
    __syncthreads();
    buf ^= 1;
  }

  for (int mt = 0; mt < 2; ++mt)
    for (int nt = 0; nt < 2; ++nt) {
      int gn = n0 + nw * 32 + nt * 16 + lm;
      float cb = bias[gn];
      for (int r = 0; r < 8; ++r) {
        int gm = m0 + mw * 32 + mt * 16 + lh * 8 + r;
        int b = gm / HW_;
        float v = acc[mt][nt][r] + cb;
        v = v * cm[b * C_ + gn] * sm[gm];
        H3[(size_t)gm * C_ + gn] = (__bf16)v;
      }
    }
}

// ---------------- coordinate attention ----------------

// XHW[b][pos][c]: pos<56 -> mean over x at y=pos; pos>=56 -> mean over y at x=pos-56
__global__ __launch_bounds__(256) void hw_means_k(const __bf16* __restrict__ H3,
                                                  float* __restrict__ XHW) {
  int b = blockIdx.y, pos = blockIdx.x;
  for (int c = threadIdx.x; c < C_; c += 256) {
    float s = 0.f;
    if (pos < 56) {
      const __bf16* r = H3 + ((size_t)b * HW_ + pos * 56) * C_;
      for (int x = 0; x < 56; ++x) s += (float)r[(size_t)x * C_ + c];
    } else {
      const __bf16* r = H3 + ((size_t)b * HW_ + (pos - 56)) * C_;
      for (int y = 0; y < 56; ++y) s += (float)r[(size_t)y * 56 * C_ + c];
    }
    XHW[((size_t)b * 112 + pos) * C_ + c] = s * (1.f / 56.f);
  }
}

// y = hswish(conv1x1 512->16)
__global__ __launch_bounds__(256) void ca_mid_k(const float* __restrict__ XHW,
                                                const float* __restrict__ W1,
                                                const float* __restrict__ B1,
                                                float* __restrict__ Y1) {
  int pos = blockIdx.x;                       // 112
  int b = threadIdx.x >> 4, mip = threadIdx.x & 15;
  const float* xv = XHW + ((size_t)b * 112 + pos) * C_;
  const float* w = W1 + (size_t)mip * C_;
  float s = B1[mip];
  for (int c = 0; c < C_; ++c) s += xv[c] * w[c];
  float r6 = fminf(fmaxf(s + 3.f, 0.f), 6.f);
  Y1[((size_t)b * 112 + pos) * 16 + mip] = s * r6 * (1.f / 6.f);
}

// ah[b][c][y], aw[b][c][x] = sigmoid(16-dot)
__global__ void ca_gates_k(const float* __restrict__ Y1, const float* __restrict__ WH,
                           const float* __restrict__ BH, const float* __restrict__ WW,
                           const float* __restrict__ BW, float* __restrict__ AH,
                           float* __restrict__ AW) {
  int i = blockIdx.x * 256 + threadIdx.x;     // b*512*56
  int b = i / (C_ * 56), r = i % (C_ * 56);
  int c = r / 56, q = r % 56;
  const float* yh = Y1 + ((size_t)b * 112 + q) * 16;
  const float* yw = Y1 + ((size_t)b * 112 + 56 + q) * 16;
  float sh = BH[c], sw = BW[c];
  for (int m = 0; m < 16; ++m) {
    sh += WH[c * 16 + m] * yh[m];
    sw += WW[c * 16 + m] * yw[m];
  }
  AH[i] = 1.f / (1.f + __expf(-sh));
  AW[i] = 1.f / (1.f + __expf(-sw));
}

// out = relu(h3 * ah * aw + x), NHWC->NCHW transpose through LDS
__global__ __launch_bounds__(256) void final_k(const __bf16* __restrict__ H3,
                                               const float* __restrict__ AH,
                                               const float* __restrict__ AW,
                                               const float* __restrict__ X,
                                               float* __restrict__ OUT) {
  __shared__ __bf16 tile[56 * 256];
  int b = blockIdx.y, y = blockIdx.x, ch0 = blockIdx.z * 256;
  const __bf16* src = H3 + ((size_t)b * HW_ + y * 56) * C_ + ch0;
  for (int idx = threadIdx.x; idx < 56 * 32; idx += 256) {  // v8bf chunks
    int xx = idx >> 5, cc = (idx & 31) * 8;
    *(v8bf*)&tile[xx * 256 + cc] = *(const v8bf*)&src[(size_t)xx * C_ + cc];
  }
  __syncthreads();
  int c = ch0 + threadIdx.x;
  float a_h = AH[((size_t)b * C_ + c) * 56 + y];
  const float* awp = AW + ((size_t)b * C_ + c) * 56;
  const float* xin = X + ((size_t)b * C_ + c) * HW_ + y * 56;
  float* op = OUT + ((size_t)b * C_ + c) * HW_ + y * 56;
  for (int xx = 0; xx < 56; ++xx) {
    float v = (float)tile[xx * 256 + threadIdx.x] * a_h * awp[xx] + xin[xx];
    op[xx] = fmaxf(v, 0.f);
  }
}

// ---------------------------------------------------------------------------

extern "C" void kernel_launch(void* const* d_in, const int* in_sizes, int n_in,
                              void* d_out, int out_size, void* d_ws, size_t ws_size,
                              hipStream_t stream) {
  const float* x     = (const float*)d_in[0];
  const float* sm_w  = (const float*)d_in[1];
  const float* sm_b  = (const float*)d_in[2];
  const float* cm1_w = (const float*)d_in[3];
  const float* cm1_b = (const float*)d_in[4];
  const float* cm2_w = (const float*)d_in[5];
  const float* cm2_b = (const float*)d_in[6];
  const float* cm3_w = (const float*)d_in[7];
  const float* cm3_b = (const float*)d_in[8];
  const float* w1    = (const float*)d_in[9];
  const float* b1    = (const float*)d_in[10];
  const float* w2    = (const float*)d_in[11];
  const float* b2    = (const float*)d_in[12];
  const float* w3    = (const float*)d_in[13];
  const float* b3    = (const float*)d_in[14];
  const float* ca_w1 = (const float*)d_in[15];
  const float* ca_b1 = (const float*)d_in[16];
  const float* ca_wh = (const float*)d_in[17];
  const float* ca_bh = (const float*)d_in[18];
  const float* ca_ww = (const float*)d_in[19];
  const float* ca_bw = (const float*)d_in[20];
  (void)in_sizes; (void)n_in; (void)out_size; (void)ws_size;

  char* ws = (char*)d_ws;
  size_t off = 0;
  auto alloc = [&](size_t bytes) -> char* {
    char* p = ws + off;
    off = (off + bytes + 511) & ~(size_t)511;
    return p;
  };

  __bf16* XT  = (__bf16*)alloc((size_t)M_ * C_ * 2);             // 51.4 MB
  __bf16* H1P = (__bf16*)alloc((size_t)B_ * 58 * 58 * WID_ * 2); // 13.8 MB
  __bf16* H2  = (__bf16*)alloc((size_t)M_ * WID_ * 2);           // 12.8 MB
  __bf16* H3  = (__bf16*)alloc((size_t)M_ * C_ * 2);             // 51.4 MB
  __bf16* W1B = (__bf16*)alloc((size_t)WID_ * C_ * 2);
  __bf16* W2B = (__bf16*)alloc((size_t)9 * WID_ * WID_ * 2);
  __bf16* W3B = (__bf16*)alloc((size_t)C_ * WID_ * 2);
  float*  SM  = (float*)alloc((size_t)M_ * 4);
  float*  SMD = (float*)alloc((size_t)M_ * 4);
  float*  PX  = (float*)alloc((size_t)B_ * C_ * 4);
  float*  CM1 = (float*)alloc((size_t)B_ * WID_ * 4);
  float*  PH1 = (float*)alloc((size_t)B_ * WID_ * 4);
  float*  CM2 = (float*)alloc((size_t)B_ * WID_ * 4);
  float*  PH2 = (float*)alloc((size_t)B_ * WID_ * 4);
  float*  CM3 = (float*)alloc((size_t)B_ * C_ * 4);
  float*  XHW = (float*)alloc((size_t)B_ * 112 * C_ * 4);
  float*  Y1  = (float*)alloc((size_t)B_ * 112 * 16 * 4);
  float*  AH  = (float*)alloc((size_t)B_ * C_ * 56 * 4);
  float*  AW  = (float*)alloc((size_t)B_ * C_ * 56 * 4);

  // prep (independent)
  cvt_bf16_k<<<(WID_ * C_) / 256, 256, 0, stream>>>(w1, W1B, WID_ * C_);
  cvt_bf16_k<<<(C_ * WID_) / 256, 256, 0, stream>>>(w3, W3B, C_ * WID_);
  perm_w2_k<<<(9 * WID_ * WID_) / 256, 256, 0, stream>>>(w2, W2B);
  transpose_cvt_k<<<dim3(98, 16, 16), 256, 0, stream>>>(x, XT);
  {
    int words = (B_ * 58 * 58 * WID_ * 2) / 4;
    zero_k<<<(words + 255) / 256, 256, 0, stream>>>((unsigned int*)H1P, words);
  }
  pool_x_k<<<(B_ * C_) / 8, 256, 0, stream>>>(x, PX);

  // masks
  smask_k<<<M_ / 256, 256, 0, stream>>>(XT, sm_w, sm_b, SM);
  dilate_k<<<M_ / 256, 256, 0, stream>>>(SM, SMD);
  cmask_k<<<(B_ * WID_) / 256, 256, 0, stream>>>(PX, cm1_w, cm1_b, CM1, WID_, C_);

  // conv1 (WMMA + TDM)
  gemm1_k<<<dim3(WID_ / 64, M_ / 128), 256, 0, stream>>>(XT, W1B, b1, CM1, SMD, H1P);
  pool_h1_k<<<B_, 128, 0, stream>>>(H1P, PH1);
  cmask_k<<<(B_ * WID_) / 256, 256, 0, stream>>>(PH1, cm2_w, cm2_b, CM2, WID_, WID_);

  // conv2 3x3 (WMMA implicit GEMM)
  conv3x3_k<<<dim3(HW_ / 64, B_), 256, 0, stream>>>(H1P, W2B, b2, CM2, SM, H2);
  pool_h2_k<<<B_, 128, 0, stream>>>(H2, PH2);
  cmask_k<<<(B_ * C_) / 256, 256, 0, stream>>>(PH2, cm3_w, cm3_b, CM3, C_, WID_);

  // conv3 (WMMA + TDM)
  gemm3_k<<<dim3(C_ / 64, M_ / 128), 256, 0, stream>>>(H2, W3B, b3, CM3, SM, H3);

  // coordinate attention + residual
  hw_means_k<<<dim3(112, B_), 256, 0, stream>>>(H3, XHW);
  ca_mid_k<<<112, 256, 0, stream>>>(XHW, ca_w1, ca_b1, Y1);
  ca_gates_k<<<(B_ * C_ * 56) / 256, 256, 0, stream>>>(Y1, ca_wh, ca_bh, ca_ww, ca_bw,
                                                       AH, AW);
  final_k<<<dim3(56, B_, 2), 256, 0, stream>>>(H3, AH, AW, x, (float*)d_out);
}
